// selfAttentionHeadMetric_31799937859873
// MI455X (gfx1250) — matrix operand order repaired and checked by reading, not compiled, not run on hardware
//
#include <hip/hip_runtime.h>
#include <hip/hip_bf16.h>

// ---------------------------------------------------------------------------
// selfAttentionHeadMetric for MI455X (gfx1250, wave32, WMMA)
//   v = (exp-kernel attention) -> 3-layer MLP with BatchNorm + leaky ReLU
// All GEMMs via v_wmma_f32_16x16x32_bf16 (bf16 in, f32 accumulate).
// f32->bf16 converts via native v_fma_mixlo/mixhi_bf16 (VOP3P 62/63).
// Working set (~30 MB) is L2-resident -> bottleneck is VALU (exp + cvt) and
// WMMA issue, so we minimize per-element VALU; no TDM needed.
// ---------------------------------------------------------------------------

typedef __attribute__((ext_vector_type(16))) __bf16 v16bf;
typedef __attribute__((ext_vector_type(8)))  float  v8f;

union BF16Tile {
    v16bf v;
    unsigned short s[16];
    unsigned int   w[8];
    uint4          q[2];
};

// Single f32 -> bf16 (RNE) via v_fma_mixlo_bf16 (sources f32 when op_sel=0).
__device__ __forceinline__ unsigned short f2bf(float f) {
    unsigned int d;
    asm("v_fma_mixlo_bf16 %0, %1, 1.0, 0" : "=v"(d) : "v"(f));
    return (unsigned short)d;
}

// Packed f32x2 -> bf16x2 (low = a, high = b): mixlo then mixhi into same dword.
__device__ __forceinline__ unsigned int pkbf(float a, float b) {
    unsigned int d;
    asm("v_fma_mixlo_bf16 %0, %1, 1.0, 0\n\t"
        "v_fma_mixhi_bf16 %0, %2, 1.0, 0"
        : "=&v"(d)
        : "v"(a), "v"(b));
    return d;
}

#define Bc 16
#define Nc 512
#define NIc 32
#define NHc 8           // heads
#define Mrows 8192      // B*N
#define EPSc 1e-5f
#define SLOPEc 0.05f

// ---------------------------------------------------------------------------
// Kernel 0: f32 -> bf16 conversion, two elements per thread (packed store).
// ---------------------------------------------------------------------------
__global__ void cvt_kernel(const float* __restrict__ src,
                           unsigned int* __restrict__ dst, int npairs) {
    for (int i = blockIdx.x * blockDim.x + threadIdx.x; i < npairs;
         i += gridDim.x * blockDim.x)
        dst[i] = pkbf(src[2 * i], src[2 * i + 1]);
}

// ---------------------------------------------------------------------------
// Kernel 1: attention.  Grid = 512 blocks (16 batches x 32 sub-blocks),
// 256 threads = 8 waves.  Each wave: one (head, 16-row) tile of temp.
// D tile (16x32 bf16) built per-lane in the WMMA A layout; B from LDS (x^T).
// pos kept SoA so each lane's 8-consecutive-m groups load as ds_load_b128.
// ---------------------------------------------------------------------------
__global__ void attn_kernel(const float* __restrict__ x,
                            const float* __restrict__ Q,
                            unsigned short* __restrict__ vout) {
    __shared__ __align__(16) unsigned short xT[NIc][Nc];  // 32 KB, bf16, [i][m]
    __shared__ __align__(16) float posX[Nc];
    __shared__ __align__(16) float posY[Nc];
    __shared__ __align__(16) float posZ[Nc];

    const int b   = blockIdx.x >> 5;       // batch
    const int sub = blockIdx.x & 31;
    const int tid = threadIdx.x;

    // Stage x[b] (512x32 f32) into LDS: transposed bf16 + SoA positions.
    for (int i = tid; i < Nc * NIc; i += 256) {
        int m = i >> 5, c = i & 31;
        float val = x[(size_t)(b * Nc + m) * NIc + c];
        xT[c][m] = f2bf(val);
        if (c == 0) posX[m] = val;
        else if (c == 1) posY[m] = val;
        else if (c == 2) posZ[m] = val;
    }
    __syncthreads();

    const int wave    = tid >> 5;
    const int lane    = tid & 31;
    const int laneHi  = lane >> 4;          // 0: lanes 0-15, 1: lanes 16-31
    const int lanelo  = lane & 15;
    const int waveJob = sub * 8 + wave;     // 0..255
    const int h  = waveJob >> 5;            // 0..7
    const int nt = waveJob & 31;            // 0..31 (16-row tile)

    const int row = nt * 16 + lanelo;       // A-matrix row M for this lane
    // t = p_n - Q_h, folded once.
    const float tnx = posX[row] - Q[h * 3 + 0];
    const float tny = posY[row] - Q[h * 3 + 1];
    const float tnz = posZ[row] - Q[h * 3 + 2];

    v8f c0 = {};   // output cols 0..15
    v8f c1 = {};   // output cols 16..31

    for (int m0 = 0; m0 < Nc; m0 += 32) {
        const int mb = m0 + laneHi * 8;     // this lane's first m
        // Wide SoA pos loads: e=0..7 -> m=mb..mb+7 ; e=8..15 -> m=mb+16..mb+23
        union { float4 v[4]; float f[16]; } PX, PY, PZ;
        {
            const float4* pX = reinterpret_cast<const float4*>(&posX[mb]);
            const float4* pY = reinterpret_cast<const float4*>(&posY[mb]);
            const float4* pZ = reinterpret_cast<const float4*>(&posZ[mb]);
            PX.v[0] = pX[0]; PX.v[1] = pX[1]; PX.v[2] = pX[4]; PX.v[3] = pX[5];
            PY.v[0] = pY[0]; PY.v[1] = pY[1]; PY.v[2] = pY[4]; PY.v[3] = pY[5];
            PZ.v[0] = pZ[0]; PZ.v[1] = pZ[1]; PZ.v[2] = pZ[4]; PZ.v[3] = pZ[5];
        }
        // A tile: D = exp(-||p_n - p_m - Q||^2) straight into bf16 A layout.
        BF16Tile a;
#pragma unroll
        for (int e = 0; e < 16; e += 2) {
            float dx0 = tnx - PX.f[e],     dy0 = tny - PY.f[e],     dz0 = tnz - PZ.f[e];
            float dx1 = tnx - PX.f[e + 1], dy1 = tny - PY.f[e + 1], dz1 = tnz - PZ.f[e + 1];
            float e0 = __expf(-(dx0 * dx0 + dy0 * dy0 + dz0 * dz0));
            float e1 = __expf(-(dx1 * dx1 + dy1 * dy1 + dz1 * dz1));
            a.w[e >> 1] = pkbf(e0, e1);
        }
        // B tiles: x[b, m0+k, i].  Lane = column i, k = kh8+0..7 and kh8+16..23.
        BF16Tile b0, b1;
        {
            int kh = laneHi * 8;
            const uint4* p0 = reinterpret_cast<const uint4*>(&xT[lanelo][m0 + kh]);
            b0.q[0] = p0[0];
            b0.q[1] = p0[2];
            const uint4* p1 = reinterpret_cast<const uint4*>(&xT[16 + lanelo][m0 + kh]);
            b1.q[0] = p1[0];
            b1.q[1] = p1[2];
        }
        c0 = __builtin_amdgcn_wmma_f32_16x16x32_bf16(false, a.v, false, b0.v,
                                                     (short)0, c0, false, false);
        c1 = __builtin_amdgcn_wmma_f32_16x16x32_bf16(false, a.v, false, b1.v,
                                                     (short)0, c1, false, false);
    }

    // Epilogue: temp[..., :3] -= pos ; store v[b, n, h*32 + i] as bf16.
    // Branch-free: unconditional in-range LDS read + select (v_cndmask).
    const float* pp = (lanelo == 0) ? posX : (lanelo == 1) ? posY : posZ;
    const bool hasP = lanelo < 3;
#pragma unroll
    for (int r = 0; r < 8; ++r) {
        int orow  = nt * 16 + r + laneHi * 8;                  // C layout row
        size_t gb = (size_t)(b * Nc + orow) * (NIc * NHc) + h * NIc;
        float pc = pp[orow];
        float v0 = c0[r] - (hasP ? pc : 0.f);
        vout[gb + lanelo]      = f2bf(v0);
        vout[gb + 16 + lanelo] = f2bf(c1[r]);
    }
}

// ---------------------------------------------------------------------------
// Kernel 2: generic bf16 GEMM  Y[M,Nout] = A[M,K] * W[Nout,K]^T + bias
// Wave job = (16-row tile) x (64-col strip) -> 4 WMMAs per k-step sharing A.
// ---------------------------------------------------------------------------
__global__ void gemm_kernel(const unsigned short* __restrict__ A,
                            const unsigned short* __restrict__ W,
                            const float* __restrict__ bias,
                            float* __restrict__ Y,
                            int M, int K, int Nout) {
    const int tid    = threadIdx.x;
    const int lane   = tid & 31;
    const int laneHi = lane >> 4;
    const int lanelo = lane & 15;
    const int job    = blockIdx.x * 8 + (tid >> 5);
    const int colStrips = Nout >> 6;
    const int rowTile = job / colStrips;
    const int colBase = (job % colStrips) << 6;
    const int arow = rowTile * 16 + lanelo;

    v8f acc[4] = {v8f{}, v8f{}, v8f{}, v8f{}};

    for (int k0 = 0; k0 < K; k0 += 32) {
        BF16Tile a;
        const uint4* pA = reinterpret_cast<const uint4*>(
            A + (size_t)arow * K + k0 + laneHi * 8);
        a.q[0] = pA[0];   // k = k0+kh8 + 0..7
        a.q[1] = pA[2];   // k = k0+kh8 + 16..23
#pragma unroll
        for (int t = 0; t < 4; ++t) {
            int n = colBase + t * 16 + lanelo;
            const uint4* pB = reinterpret_cast<const uint4*>(
                W + (size_t)n * K + k0 + laneHi * 8);
            BF16Tile bm;
            bm.q[0] = pB[0];
            bm.q[1] = pB[2];
            acc[t] = __builtin_amdgcn_wmma_f32_16x16x32_bf16(
                false, a.v, false, bm.v, (short)0, acc[t], false, false);
        }
    }

#pragma unroll
    for (int t = 0; t < 4; ++t) {
        int col  = colBase + t * 16 + lanelo;
        float bv = bias[col];
#pragma unroll
        for (int r = 0; r < 8; ++r) {
            int orow = rowTile * 16 + r + laneHi * 8;
            Y[(size_t)orow * Nout + col] = acc[t][r] + bv;
        }
    }
}

// ---------------------------------------------------------------------------
// Kernel 3: per-column BatchNorm stats -> scale/shift (one block per column)
// ---------------------------------------------------------------------------
__global__ void colstats_kernel(const float* __restrict__ Y,
                                const float* __restrict__ g,
                                const float* __restrict__ bt,
                                float* __restrict__ scale,
                                float* __restrict__ shift,
                                int M, int Nout) {
    __shared__ float ssum[256];
    __shared__ float ssq[256];
    const int col = blockIdx.x;
    const int tid = threadIdx.x;
    float s = 0.f, s2 = 0.f;
    for (int r = tid; r < M; r += 256) {
        float v = Y[(size_t)r * Nout + col];
        s += v;
        s2 += v * v;
    }
    ssum[tid] = s;
    ssq[tid]  = s2;
    __syncthreads();
    for (int off = 128; off > 0; off >>= 1) {
        if (tid < off) {
            ssum[tid] += ssum[tid + off];
            ssq[tid]  += ssq[tid + off];
        }
        __syncthreads();
    }
    if (tid == 0) {
        float mean = ssum[0] / (float)M;
        float var  = ssq[0] / (float)M - mean * mean;
        float rstd = rsqrtf(var + EPSc);
        float sc   = g[col] * rstd;
        scale[col] = sc;
        shift[col] = bt[col] - mean * sc;
    }
}

// ---------------------------------------------------------------------------
// Kernel 4: normalize (+ optional leaky ReLU); two elements per thread so the
// bf16 path emits packed 32-bit stores.  Nout is even -> pairs never wrap.
// ---------------------------------------------------------------------------
__global__ void bnact_kernel(const float* __restrict__ Y,
                             const float* __restrict__ scale,
                             const float* __restrict__ shift,
                             unsigned int* __restrict__ abf,
                             float2* __restrict__ fout,
                             int totalPairs, int Nout, int leaky) {
    for (int i = blockIdx.x * blockDim.x + threadIdx.x; i < totalPairs;
         i += gridDim.x * blockDim.x) {
        int idx  = 2 * i;
        int col0 = idx & (Nout - 1);        // Nout is a power of two
        int col1 = col0 + 1;
        float z0 = Y[idx]     * scale[col0] + shift[col0];
        float z1 = Y[idx + 1] * scale[col1] + shift[col1];
        if (leaky) {
            z0 = z0 > 0.f ? z0 : z0 * SLOPEc;
            z1 = z1 > 0.f ? z1 : z1 * SLOPEc;
        }
        if (fout) fout[i] = make_float2(z0, z1);
        else      abf[i]  = pkbf(z0, z1);
    }
}

// ---------------------------------------------------------------------------
// Host launcher
// ---------------------------------------------------------------------------
extern "C" void kernel_launch(void* const* d_in, const int* in_sizes, int n_in,
                              void* d_out, int out_size, void* d_ws, size_t ws_size,
                              hipStream_t stream) {
    const float* x   = (const float*)d_in[0];
    const float* Q   = (const float*)d_in[1];
    const float* W1  = (const float*)d_in[2];
    const float* b1  = (const float*)d_in[3];
    const float* g1  = (const float*)d_in[4];
    const float* bt1 = (const float*)d_in[5];
    const float* W2  = (const float*)d_in[6];
    const float* b2  = (const float*)d_in[7];
    const float* g2  = (const float*)d_in[8];
    const float* bt2 = (const float*)d_in[9];
    const float* W3  = (const float*)d_in[10];
    const float* b3  = (const float*)d_in[11];
    const float* g3  = (const float*)d_in[12];
    const float* bt3 = (const float*)d_in[13];
    float* out = (float*)d_out;

    // Workspace carve-up (all 256B aligned)
    char* ws = (char*)d_ws;
    auto carve = [&](size_t bytes) {
        char* p = ws;
        ws += (bytes + 255) & ~(size_t)255;
        return p;
    };
    unsigned short* W1b  = (unsigned short*)carve((size_t)512 * 256 * 2);
    unsigned short* W2b  = (unsigned short*)carve((size_t)512 * 512 * 2);
    unsigned short* W3b  = (unsigned short*)carve((size_t)64 * 512 * 2);
    unsigned short* vbf  = (unsigned short*)carve((size_t)Mrows * 256 * 2);
    unsigned short* act  = (unsigned short*)carve((size_t)Mrows * 512 * 2);
    float*          Y    = (float*)carve((size_t)Mrows * 512 * 4);
    float*          scl  = (float*)carve(512 * 4);
    float*          shf  = (float*)carve(512 * 4);

    // Weights -> bf16 (packed converts)
    cvt_kernel<<<64, 256, 0, stream>>>(W1, (unsigned int*)W1b, 512 * 256 / 2);
    cvt_kernel<<<128, 256, 0, stream>>>(W2, (unsigned int*)W2b, 512 * 512 / 2);
    cvt_kernel<<<32, 256, 0, stream>>>(W3, (unsigned int*)W3b, 64 * 512 / 2);

    // Attention: v[8192, 256] (bf16)
    attn_kernel<<<512, 256, 0, stream>>>(x, Q, vbf);

    // Layer 1: y1 = v @ W1^T + b1 ; BN ; leaky
    gemm_kernel<<<512, 256, 0, stream>>>(vbf, W1b, b1, Y, Mrows, 256, 512);
    colstats_kernel<<<512, 256, 0, stream>>>(Y, g1, bt1, scl, shf, Mrows, 512);
    bnact_kernel<<<2048, 256, 0, stream>>>(Y, scl, shf, (unsigned int*)act,
                                           nullptr, Mrows * 512 / 2, 512, 1);

    // Layer 2: y2 = a1 @ W2^T + b2 ; BN ; leaky
    gemm_kernel<<<512, 256, 0, stream>>>(act, W2b, b2, Y, Mrows, 512, 512);
    colstats_kernel<<<512, 256, 0, stream>>>(Y, g2, bt2, scl, shf, Mrows, 512);
    bnact_kernel<<<2048, 256, 0, stream>>>(Y, scl, shf, (unsigned int*)act,
                                           nullptr, Mrows * 512 / 2, 512, 1);

    // Layer 3: y3 = a2 @ W3^T + b3 ; BN ; f32 output
    gemm_kernel<<<64, 256, 0, stream>>>(act, W3b, b3, Y, Mrows, 512, 64);
    colstats_kernel<<<64, 256, 0, stream>>>(Y, g3, bt3, scl, shf, Mrows, 64);
    bnact_kernel<<<1024, 256, 0, stream>>>(Y, scl, shf, nullptr, (float2*)out,
                                           Mrows * 64 / 2, 64, 0);
}
